// Attention_35923106463893
// MI455X (gfx1250) — matrix operand order repaired and checked by reading, not compiled, not run on hardware
//
#include <hip/hip_runtime.h>

typedef unsigned short u16;
typedef __attribute__((ext_vector_type(16))) __bf16 v16bf;
typedef __attribute__((ext_vector_type(8)))  float  v8f;
typedef __attribute__((ext_vector_type(4)))  unsigned int v4u;
typedef __attribute__((ext_vector_type(8)))  int v8i;
typedef __attribute__((ext_vector_type(4)))  int v4i;

struct alignas(16) U4 { unsigned int x, y, z, w; };
union FragU { U4 u[2]; v16bf v; };

#define B_SZ     8
#define N_TOK    1024
#define D_MODEL  768
#define N_HEAD   12
#define HEAD_D   64
#define QK_SCALE 0.125f              // 64^-0.5
#define M_ROWS   (B_SZ * N_TOK)      // 8192
#define QKV_COLS (3 * D_MODEL)       // 2304

#if __has_builtin(__builtin_amdgcn_tensor_load_to_lds)
#define HAVE_TDM 1
#endif

__device__ __forceinline__ u16 f2bf(float f) {
  unsigned int u = __float_as_uint(f);
  u += 0x7FFFu + ((u >> 16) & 1u);   // round-to-nearest-even
  return (u16)(u >> 16);
}

__device__ __forceinline__ v8f zero8() {
  v8f z;
#pragma unroll
  for (int i = 0; i < 8; ++i) z[i] = 0.0f;
  return z;
}

__device__ __forceinline__ v8f wmma_bf16(v16bf a, v16bf b, v8f c) {
  return __builtin_amdgcn_wmma_f32_16x16x32_bf16(false, a, false, b, (short)0, c,
                                                 false, false);
}

__device__ __forceinline__ float redmax16(float v) {
  v = fmaxf(v, __shfl_xor(v, 1, 32));
  v = fmaxf(v, __shfl_xor(v, 2, 32));
  v = fmaxf(v, __shfl_xor(v, 4, 32));
  v = fmaxf(v, __shfl_xor(v, 8, 32));
  return v;
}
__device__ __forceinline__ float redsum16(float v) {
  v += __shfl_xor(v, 1, 32);
  v += __shfl_xor(v, 2, 32);
  v += __shfl_xor(v, 4, 32);
  v += __shfl_xor(v, 8, 32);
  return v;
}

// ---------------- TDM helpers ----------------
__device__ __forceinline__ void tdm_wait0() {
#if __has_builtin(__builtin_amdgcn_s_wait_tensorcnt)
  __builtin_amdgcn_s_wait_tensorcnt(0);
#else
  asm volatile("s_wait_tensorcnt 0x0" ::: "memory");
#endif
}

#ifdef HAVE_TDM
// 2D tile DMA: global (row-major, stride0 elems/row, 2-byte elems) -> LDS.
// D# per CDNA5 ISA ch.8: group0 = {count|flags, lds_addr, global_addr, type=2},
// group1 = {data_size, tensor dims, tile dims, dim0 stride}. Groups 2/3 zero
// (tile_dim2/3 = 0 -> unused).
__device__ __forceinline__ void tdm_load_2d(unsigned lds_off, const void* gptr,
                                            unsigned tile0, unsigned tile1,
                                            unsigned td0, unsigned td1,
                                            unsigned stride0) {
  unsigned long long ga = (unsigned long long)(uintptr_t)gptr;
  v4u g0 = { 1u,                                  // count=1, user mode, no gather
             lds_off,                             // lds_addr (bytes)
             (unsigned)ga,                        // global_addr[31:0]
             (unsigned)(((ga >> 32) & 0x01ffffffu) | 0x80000000u) }; // [56:32]|type=2
  v8i g1;
  g1[0] = (int)(1u << 16);                              // data_size=1 (2 bytes)
  g1[1] = (int)((td0 & 0xffffu) << 16);                 // tensor_dim0[15:0]
  g1[2] = (int)(((td0 >> 16) & 0xffffu) | ((td1 & 0xffffu) << 16));
  g1[3] = (int)(((td1 >> 16) & 0xffffu) | ((tile0 & 0xffffu) << 16));
  g1[4] = (int)(tile1 & 0xffffu);                       // tile_dim1; tile_dim2=0
  g1[5] = (int)stride0;                                 // dim0_stride[31:0]
  g1[6] = 0;                                            // stride[47:32], s1[15:0]
  g1[7] = 0;
  v4i gz = {0, 0, 0, 0};
#if defined(__clang_major__) && (__clang_major__ >= 23)
  v8i gz8 = {0, 0, 0, 0, 0, 0, 0, 0};
  __builtin_amdgcn_tensor_load_to_lds(g0, g1, gz, gz, gz8, 0);
#else
  __builtin_amdgcn_tensor_load_to_lds(g0, g1, gz, gz, 0);
#endif
}
#endif

// ---------------- fp32 -> bf16 convert ----------------
__global__ void cvt_f32_bf16(const float* __restrict__ s, u16* __restrict__ d, int n) {
  int i = blockIdx.x * blockDim.x + threadIdx.x;
  int stride = gridDim.x * blockDim.x;
  for (; i < n; i += stride) d[i] = f2bf(s[i]);
}

// ---------------- bf16 GEMM (shared core, two epilogues) ----------------
// MODE 0: QKV projection. C[row,col] *= gate[row]; Q scaled by QK_SCALE;
//         scatter to Q[B,H,N,64], K[B,H,N,64], VT[B,H,64,N] (bf16).
// MODE 1: output projection. out[row,col] = C + bias[col] (fp32).
// Block: 256 threads = 8 waves as 4(M)x2(N); block tile 128x64; K step 32.
// A tile (128x32 bf16, 8KB) staged per k-step by the Tensor Data Mover
// (issued + tensorcnt-waited by wave 0, published by the block barrier)
// while the other waves transpose-stage the B tile.
template <int MODE>
__global__ void __launch_bounds__(256)
gemm_bf16_kernel(const u16* __restrict__ A, const u16* __restrict__ Bm,
                 int Ncols, int K, const float* __restrict__ gb,
                 u16* __restrict__ qb, u16* __restrict__ kb, u16* __restrict__ vtb,
                 float* __restrict__ outf) {
  __shared__ alignas(16) u16 sA[128 * 32];   // [row][k] row-major
  __shared__ alignas(16) u16 sB[64 * 32];    // [n][k]  (B transposed)

  const int tid = threadIdx.x, lane = tid & 31, wid = tid >> 5;
  const int half = lane >> 4, l15 = lane & 15;
  const int waveM = wid >> 1, waveN = wid & 1;
  const int rowBase = blockIdx.y * 128;
  const int colBase = blockIdx.x * 64;

  v8f acc[2][2];
#pragma unroll
  for (int i = 0; i < 2; ++i)
#pragma unroll
    for (int j = 0; j < 2; ++j) acc[i][j] = zero8();

  for (int k0 = 0; k0 < K; k0 += 32) {
#ifdef HAVE_TDM
    if (wid == 0) {  // TDM DMA of A tile; wave0 owns the TENSORcnt wait
      tdm_load_2d((unsigned)(uintptr_t)(void*)sA,
                  A + (size_t)rowBase * K + k0,
                  /*tile0*/ 32u, /*tile1*/ 128u,
                  /*tensor_d0*/ (unsigned)K, /*tensor_d1*/ (unsigned)M_ROWS,
                  /*stride0*/ (unsigned)K);
      tdm_wait0();
    }
#else
    { // manual staging fallback: 128x32 bf16, 32B per thread
      int r = tid >> 1, kh = (tid & 1) * 16;
      const U4* gp = (const U4*)(A + (size_t)(rowBase + r) * K + k0 + kh);
      U4 a0 = gp[0], a1 = gp[1];
      *(U4*)(sA + r * 32 + kh)     = a0;
      *(U4*)(sA + r * 32 + kh + 8) = a1;
    }
#endif
    { // stage B tile transposed: global [k][n] -> LDS [n][k]
      int kk = tid >> 3, n0 = (tid & 7) * 8;
      U4 bv = *(const U4*)(Bm + (size_t)(k0 + kk) * Ncols + colBase + n0);
      const u16* bp = (const u16*)&bv;
#pragma unroll
      for (int i = 0; i < 8; ++i) sB[(n0 + i) * 32 + kk] = bp[i];
    }
    __syncthreads();

    v16bf afr[2], bfr[2];
#pragma unroll
    for (int i = 0; i < 2; ++i) {  // A frag: lanes0-15 K=h*8..+7 & 16+h*8..+7
      int mr = waveM * 32 + i * 16 + l15;
      FragU f;
      f.u[0] = *(const U4*)(sA + mr * 32 + half * 8);
      f.u[1] = *(const U4*)(sA + mr * 32 + 16 + half * 8);
      afr[i] = f.v;
    }
#pragma unroll
    for (int j = 0; j < 2; ++j) {  // B frag: K=h*8..+7 & 16+h*8..+7
      int nc = waveN * 32 + j * 16 + l15;
      FragU f;
      f.u[0] = *(const U4*)(sB + nc * 32 + half * 8);
      f.u[1] = *(const U4*)(sB + nc * 32 + 16 + half * 8);
      bfr[j] = f.v;
    }
#pragma unroll
    for (int i = 0; i < 2; ++i)
#pragma unroll
      for (int j = 0; j < 2; ++j) acc[i][j] = wmma_bf16(afr[i], bfr[j], acc[i][j]);
    __syncthreads();
  }

  // C layout: VGPR r -> (M = r + 8*half, N = l15) within each 16x16 frag
#pragma unroll
  for (int i = 0; i < 2; ++i) {
#pragma unroll
    for (int j = 0; j < 2; ++j) {
#pragma unroll
      for (int r = 0; r < 8; ++r) {
        int row = rowBase + waveM * 32 + i * 16 + r + 8 * half;
        int col = colBase + waveN * 32 + j * 16 + l15;
        float v = acc[i][j][r];
        if (MODE == 0) {
          v *= gb[row];  // per-token gate
          int which = col / D_MODEL;
          int rem = col - which * D_MODEL;
          int h = rem >> 6, d = rem & 63;
          int b = row >> 10, n = row & (N_TOK - 1);
          if (which == 0)
            qb[(((size_t)(b * N_HEAD + h) * N_TOK) + n) * HEAD_D + d] = f2bf(v * QK_SCALE);
          else if (which == 1)
            kb[(((size_t)(b * N_HEAD + h) * N_TOK) + n) * HEAD_D + d] = f2bf(v);
          else
            vtb[((size_t)(b * N_HEAD + h) * HEAD_D + d) * N_TOK + n] = f2bf(v);
        } else {
          outf[(size_t)row * D_MODEL + col] = v + gb[col];
        }
      }
    }
  }
}

// ---------------- flash attention ----------------
// grid: (N_TOK/128, B*H); block 256 = 8 waves, wave owns 16 query rows.
__global__ void __launch_bounds__(256)
attn_flash_kernel(const u16* __restrict__ Q, const u16* __restrict__ K,
                  const u16* __restrict__ VT, u16* __restrict__ AO) {
  __shared__ alignas(16) u16 sP[8][16 * 32];  // per-wave P relayout slab

  const int tid = threadIdx.x, lane = tid & 31, wid = tid >> 5;
  const int half = lane >> 4, l15 = lane & 15;
  const int bh = blockIdx.y;
  const int b = bh / N_HEAD, h = bh - b * N_HEAD;
  const int qrow0 = blockIdx.x * 128 + wid * 16;
  const u16* Qp = Q + (size_t)bh * N_TOK * HEAD_D;
  const u16* Kp = K + (size_t)bh * N_TOK * HEAD_D;
  const u16* Vp = VT + (size_t)bh * HEAD_D * N_TOK;
  u16* myP = sP[wid];

  v16bf aq[2];  // Q tile 16x64 as two A-frags (d 0..31, 32..63)
  {
    const u16* qr = Qp + (size_t)(qrow0 + l15) * HEAD_D;
#pragma unroll
    for (int t = 0; t < 2; ++t) {
      FragU f;
      f.u[0] = *(const U4*)(qr + t * 32 + half * 8);
      f.u[1] = *(const U4*)(qr + t * 32 + 16 + half * 8);
      aq[t] = f.v;
    }
  }

  v8f o0 = zero8(), o1 = zero8(), o2 = zero8(), o3 = zero8();
  float m8[8], l8[8];
#pragma unroll
  for (int r = 0; r < 8; ++r) { m8[r] = -3.0e38f; l8[r] = 0.0f; }

  for (int j = 0; j < N_TOK; j += 32) {
    // S = Q K^T for 32 keys: B-frag column n = key, K-dim = head dim (contig in K rows)
    v8f s0 = zero8(), s1 = zero8();
#pragma unroll
    for (int ct = 0; ct < 2; ++ct) {
      const u16* kr = Kp + (size_t)(j + ct * 16 + l15) * HEAD_D;
#pragma unroll
      for (int t = 0; t < 2; ++t) {
        FragU f;
        f.u[0] = *(const U4*)(kr + t * 32 + half * 8);
        f.u[1] = *(const U4*)(kr + t * 32 + 16 + half * 8);
        if (ct == 0) s0 = wmma_bf16(aq[t], f.v, s0);
        else         s1 = wmma_bf16(aq[t], f.v, s1);
      }
    }
    // online softmax (rows live across 16-lane halves in one VGPR slot)
    float p0[8], p1[8];
#pragma unroll
    for (int r = 0; r < 8; ++r) {
      float mx = redmax16(fmaxf(s0[r], s1[r]));
      float mnew = fmaxf(m8[r], mx);
      float al = __expf(m8[r] - mnew);
      float e0 = __expf(s0[r] - mnew);
      float e1 = __expf(s1[r] - mnew);
      float rs = redsum16(e0 + e1);
      l8[r] = l8[r] * al + rs;
      m8[r] = mnew;
      p0[r] = e0; p1[r] = e1;
      o0[r] *= al; o1[r] *= al; o2[r] *= al; o3[r] *= al;
    }
    // relayout P (C layout) -> A frag via per-wave LDS slab [m][k], k=32
#pragma unroll
    for (int r = 0; r < 8; ++r) {
      int row = r + 8 * half;
      myP[row * 32 + l15]      = f2bf(p0[r]);
      myP[row * 32 + 16 + l15] = f2bf(p1[r]);
    }
    FragU fp;  // DS ops from one wave are in-order; compiler inserts dscnt waits
    fp.u[0] = *(const U4*)(myP + l15 * 32 + half * 8);
    fp.u[1] = *(const U4*)(myP + l15 * 32 + 16 + half * 8);
    v16bf ap = fp.v;
    // O += P V : B-frag column n = d (VT rows), K-dim = keys (contig in VT)
#pragma unroll
    for (int f4 = 0; f4 < 4; ++f4) {
      const u16* vr = Vp + (size_t)(f4 * 16 + l15) * N_TOK + j + half * 8;
      FragU fb;
      fb.u[0] = *(const U4*)(vr);
      fb.u[1] = *(const U4*)(vr + 16);
      if (f4 == 0)      o0 = wmma_bf16(ap, fb.v, o0);
      else if (f4 == 1) o1 = wmma_bf16(ap, fb.v, o1);
      else if (f4 == 2) o2 = wmma_bf16(ap, fb.v, o2);
      else              o3 = wmma_bf16(ap, fb.v, o3);
    }
  }
  // normalize and write attn output [B,N,D] bf16 (row-major for final GEMM)
#pragma unroll
  for (int r = 0; r < 8; ++r) {
    float inv = 1.0f / l8[r];
    int row = qrow0 + r + 8 * half;
    u16* orow = AO + ((size_t)(b * N_TOK + row)) * D_MODEL + h * HEAD_D;
    orow[0 * 16 + l15] = f2bf(o0[r] * inv);
    orow[1 * 16 + l15] = f2bf(o1[r] * inv);
    orow[2 * 16 + l15] = f2bf(o2[r] * inv);
    orow[3 * 16 + l15] = f2bf(o3[r] * inv);
  }
}

// ---------------- launch ----------------
extern "C" void kernel_launch(void* const* d_in, const int* in_sizes, int n_in,
                              void* d_out, int out_size, void* d_ws, size_t ws_size,
                              hipStream_t stream) {
  const float* x      = (const float*)d_in[0];
  const float* weight = (const float*)d_in[1];
  const float* W_qkv  = (const float*)d_in[2];
  const float* W_msa  = (const float*)d_in[3];
  const float* b_msa  = (const float*)d_in[4];
  float* out = (float*)d_out;

  const size_t nXB    = (size_t)M_ROWS * D_MODEL;                  // 6,291,456
  const size_t nWQKV  = (size_t)D_MODEL * QKV_COLS;
  const size_t nWMSA  = (size_t)D_MODEL * D_MODEL;
  const size_t nHEADS = (size_t)B_SZ * N_HEAD * N_TOK * HEAD_D;    // == nXB

  u16* xb    = (u16*)d_ws;
  u16* wqkvb = xb + nXB;
  u16* wmsab = wqkvb + nWQKV;
  u16* qb    = wmsab + nWMSA;
  u16* kb    = qb + nHEADS;
  u16* vtb   = kb + nHEADS;
  u16* ab    = vtb + nHEADS;     // total ~67.6 MB of d_ws

  cvt_f32_bf16<<<dim3((unsigned)((nXB + 255) / 256)), dim3(256), 0, stream>>>(x, xb, (int)nXB);
  cvt_f32_bf16<<<dim3((unsigned)((nWQKV + 255) / 256)), dim3(256), 0, stream>>>(W_qkv, wqkvb, (int)nWQKV);
  cvt_f32_bf16<<<dim3((unsigned)((nWMSA + 255) / 256)), dim3(256), 0, stream>>>(W_msa, wmsab, (int)nWMSA);

  gemm_bf16_kernel<0><<<dim3(QKV_COLS / 64, M_ROWS / 128), dim3(256), 0, stream>>>(
      xb, wqkvb, QKV_COLS, D_MODEL, weight, qb, kb, vtb, nullptr);

  attn_flash_kernel<<<dim3(N_TOK / 128, B_SZ * N_HEAD), dim3(256), 0, stream>>>(
      qb, kb, vtb, ab);

  gemm_bf16_kernel<1><<<dim3(D_MODEL / 64, M_ROWS / 128), dim3(256), 0, stream>>>(
      ab, wmsab, D_MODEL, D_MODEL, b_msa, nullptr, nullptr, nullptr, out);
}
